// Decoder_171798692161
// MI455X (gfx1250) — compile-verified, matrix operand order checked by reading, not run on hardware
//
#include <hip/hip_runtime.h>
#include <math.h>

// ---------------------------------------------------------------------------
// Types for CDNA5 WMMA (wave32, V_WMMA_F32_16X16X32_BF16)
// ---------------------------------------------------------------------------
typedef __bf16 bf16_t;
typedef __attribute__((ext_vector_type(16))) __bf16        v16bf;
typedef __attribute__((ext_vector_type(8)))  float         v8f;
typedef __attribute__((ext_vector_type(8)))  unsigned int  v8u;

static constexpr int cB  = 32;
static constexpr int cT  = 64;
static constexpr int cS  = 256;
static constexpr int cE  = 512;
static constexpr int cH  = 1024;
static constexpr int cV  = 32000;
static constexpr int cNH = 8;
static constexpr int cDH = 64;
static constexpr int cL  = 2;

__device__ __forceinline__ bf16_t f2bf(float f) {
  unsigned u = __builtin_bit_cast(unsigned, f);
  unsigned r = u + 0x7FFFu + ((u >> 16) & 1u);   // round-to-nearest-even
  unsigned short h = (unsigned short)(r >> 16);
  return __builtin_bit_cast(bf16_t, h);
}

// A/B fragment loader for 16-bit 16x32 operand, contiguous-K source row.
// ISA 7.12.2: VGPR v(0..3) holds K = 2v,2v+1 (+8 for lanes 16..31);
//             VGPR v(4..7) holds K = 16+2(v-4),+1 (+8 for lanes 16..31).
// Caller passes p = rowBase + k0 + half*8 (all even -> dword aligned).
__device__ __forceinline__ v16bf load_pairs(const bf16_t* p) {
  v8u u;
#pragma unroll
  for (int i = 0; i < 4; ++i) u[i] = *(const unsigned*)(p + 2 * i);
#pragma unroll
  for (int i = 0; i < 4; ++i) u[4 + i] = *(const unsigned*)(p + 16 + 2 * i);
  return __builtin_bit_cast(v16bf, u);
}

// B fragment when B is row-major [K,N] (non-transposed): gather a column.
__device__ __forceinline__ v16bf load_col(const bf16_t* Bm, int ldb, int n, int kbase) {
  v8u u;
#pragma unroll
  for (int i = 0; i < 4; ++i) {
    unsigned lo = (unsigned)__builtin_bit_cast(unsigned short, Bm[(size_t)(kbase + 2 * i) * ldb + n]);
    unsigned hi = (unsigned)__builtin_bit_cast(unsigned short, Bm[(size_t)(kbase + 2 * i + 1) * ldb + n]);
    u[i] = lo | (hi << 16);
  }
#pragma unroll
  for (int i = 0; i < 4; ++i) {
    unsigned lo = (unsigned)__builtin_bit_cast(unsigned short, Bm[(size_t)(kbase + 16 + 2 * i) * ldb + n]);
    unsigned hi = (unsigned)__builtin_bit_cast(unsigned short, Bm[(size_t)(kbase + 16 + 2 * i + 1) * ldb + n]);
    u[4 + i] = lo | (hi << 16);
  }
  return __builtin_bit_cast(v16bf, u);
}

// ---------------------------------------------------------------------------
// Register-blocked TN WMMA GEMM: C = alpha * (A[M,K] @ W[N,K]^T) + bias.
// One wave computes a 32x64 C block: 2 A-fragments x 4 B-fragments -> 8 WMMAs
// per 32-deep K step (0.75 fragment loads per WMMA vs 2.0 unblocked).
// Requires M%32==0, N%64==0, K%32==0 (true for every TN GEMM here).
// Batched over gridDim.y with two-level (b,h) strides, z = b*nh + h.
// ---------------------------------------------------------------------------
__global__ void k_wmma_gemm_tn(const bf16_t* __restrict__ A, const bf16_t* __restrict__ Bm,
                               const float* __restrict__ bias,
                               float* __restrict__ C, bf16_t* __restrict__ Cbf,
                               int M, int N, int K, int lda, int ldb, int ldc,
                               float alpha, int act, int nh,
                               long sAb, long sAh, long sBb, long sBh, long sCb, long sCh) {
  int lane = threadIdx.x & 31;
  int wv   = (int)(threadIdx.x >> 5);
  int tilesN = N >> 6;                               // 64-wide N blocks
  long tile  = (long)blockIdx.x * (blockDim.x >> 5) + wv;
  long total = (long)(M >> 5) * tilesN;              // 32-tall M blocks
  if (tile >= total) return;                         // wave-uniform branch
  int tm = (int)(tile / tilesN);
  int tn = (int)(tile % tilesN);
  int z  = blockIdx.y;
  int zb = z / nh, zh = z - zb * nh;
  A  += (size_t)zb * sAb + (size_t)zh * sAh;
  Bm += (size_t)zb * sBb + (size_t)zh * sBh;
  size_t cOff = (size_t)((long)zb * sCb + (long)zh * sCh);

  int half = lane >> 4;
  int l15  = lane & 15;
  const bf16_t* arow0 = A + (size_t)(tm * 32 + l15) * lda;
  const bf16_t* arow1 = arow0 + (size_t)16 * lda;
  const bf16_t* brow0 = Bm + (size_t)(tn * 64 + l15) * ldb;
  const size_t bstep = (size_t)16 * ldb;

  v8f acc[2][4];
#pragma unroll
  for (int i = 0; i < 2; ++i)
#pragma unroll
    for (int j = 0; j < 4; ++j)
#pragma unroll
      for (int r = 0; r < 8; ++r) acc[i][j][r] = 0.0f;

  for (int k0 = 0; k0 < K; k0 += 32) {
    int ko = k0 + half * 8;
    v16bf a0 = load_pairs(arow0 + ko);
    v16bf a1 = load_pairs(arow1 + ko);
    v16bf b0 = load_pairs(brow0 + ko);
    v16bf b1 = load_pairs(brow0 + bstep + ko);
    v16bf b2 = load_pairs(brow0 + 2 * bstep + ko);
    v16bf b3 = load_pairs(brow0 + 3 * bstep + ko);
    acc[0][0] = __builtin_amdgcn_wmma_f32_16x16x32_bf16(false, a0, false, b0, (short)0, acc[0][0], false, false);
    acc[0][1] = __builtin_amdgcn_wmma_f32_16x16x32_bf16(false, a0, false, b1, (short)0, acc[0][1], false, false);
    acc[0][2] = __builtin_amdgcn_wmma_f32_16x16x32_bf16(false, a0, false, b2, (short)0, acc[0][2], false, false);
    acc[0][3] = __builtin_amdgcn_wmma_f32_16x16x32_bf16(false, a0, false, b3, (short)0, acc[0][3], false, false);
    acc[1][0] = __builtin_amdgcn_wmma_f32_16x16x32_bf16(false, a1, false, b0, (short)0, acc[1][0], false, false);
    acc[1][1] = __builtin_amdgcn_wmma_f32_16x16x32_bf16(false, a1, false, b1, (short)0, acc[1][1], false, false);
    acc[1][2] = __builtin_amdgcn_wmma_f32_16x16x32_bf16(false, a1, false, b2, (short)0, acc[1][2], false, false);
    acc[1][3] = __builtin_amdgcn_wmma_f32_16x16x32_bf16(false, a1, false, b3, (short)0, acc[1][3], false, false);
  }

#pragma unroll
  for (int i = 0; i < 2; ++i) {
    int mBase = tm * 32 + i * 16;
#pragma unroll
    for (int j = 0; j < 4; ++j) {
      int nCol = tn * 64 + j * 16 + l15;
      float bv = bias ? bias[nCol] : 0.0f;
#pragma unroll
      for (int r = 0; r < 8; ++r) {
        int m = mBase + r + half * 8;                // ISA 7.12.2 C/D layout
        float v = acc[i][j][r] * alpha + bv;
        if (act == 1) v = tanhf(v);
        size_t idx = cOff + (size_t)m * ldc + nCol;
        if (C)   C[idx]   = v;
        if (Cbf) Cbf[idx] = f2bf(v);
      }
    }
  }
}

// ---------------------------------------------------------------------------
// Simple NN WMMA GEMM (B row-major [K,N]), 1 wave = one 16x16 tile.
// Used only for the small batched attn @ V GEMM.
// ---------------------------------------------------------------------------
__global__ void k_wmma_gemm_nn(const bf16_t* __restrict__ A, const bf16_t* __restrict__ Bm,
                               float* __restrict__ C, bf16_t* __restrict__ Cbf,
                               int M, int N, int K, int lda, int ldb, int ldc,
                               int nh,
                               long sAb, long sAh, long sBb, long sBh, long sCb, long sCh) {
  int lane = threadIdx.x & 31;
  int wv   = (int)(threadIdx.x >> 5);
  int tilesN = N >> 4;
  long tile  = (long)blockIdx.x * (blockDim.x >> 5) + wv;
  long total = (long)(M >> 4) * tilesN;
  if (tile >= total) return;
  int tm = (int)(tile / tilesN);
  int tn = (int)(tile % tilesN);
  int z  = blockIdx.y;
  int zb = z / nh, zh = z - zb * nh;
  A  += (size_t)zb * sAb + (size_t)zh * sAh;
  Bm += (size_t)zb * sBb + (size_t)zh * sBh;
  size_t cOff = (size_t)((long)zb * sCb + (long)zh * sCh);

  int half = lane >> 4;
  int l15  = lane & 15;
  const bf16_t* arow = A + (size_t)(tm * 16 + l15) * lda;

  v8f acc;
#pragma unroll
  for (int r = 0; r < 8; ++r) acc[r] = 0.0f;

  for (int k0 = 0; k0 < K; k0 += 32) {
    v16bf af  = load_pairs(arow + k0 + half * 8);
    v16bf bfr = load_col(Bm, ldb, tn * 16 + l15, k0 + half * 8);
    acc = __builtin_amdgcn_wmma_f32_16x16x32_bf16(false, af, false, bfr, (short)0, acc, false, false);
  }

  int nCol = tn * 16 + l15;
#pragma unroll
  for (int r = 0; r < 8; ++r) {
    int m = tm * 16 + r + half * 8;
    float v = acc[r];
    size_t idx = cOff + (size_t)m * ldc + nCol;
    if (C)   C[idx]   = v;
    if (Cbf) Cbf[idx] = f2bf(v);
  }
}

// ---------------------------------------------------------------------------
// Elementwise / reduction kernels
// ---------------------------------------------------------------------------
__global__ void k_f32_to_bf16(const float* __restrict__ src, bf16_t* __restrict__ dst, long n) {
  long i = (long)blockIdx.x * blockDim.x + threadIdx.x;
  if (i < n) dst[i] = f2bf(src[i]);
}

__global__ void k_embed(const int* __restrict__ ids, const float* __restrict__ W,
                        float* __restrict__ outf, bf16_t* __restrict__ outb,
                        int rows, int Ecols) {
  long i = (long)blockIdx.x * blockDim.x + threadIdx.x;
  if (i >= (long)rows * Ecols) return;
  long r = i / Ecols;
  int  e = (int)(i - r * Ecols);
  float v = W[(size_t)ids[r] * Ecols + e];
  outf[i] = v;
  outb[i] = f2bf(v);
}

__global__ void k_concat2(const float* __restrict__ a, int wa,
                          const float* __restrict__ b, int wb,
                          bf16_t* __restrict__ dst, int rows) {
  int w = wa + wb;
  long i = (long)blockIdx.x * blockDim.x + threadIdx.x;
  if (i >= (long)rows * w) return;
  long r = i / w;
  int  c = (int)(i - r * w);
  float v = (c < wa) ? a[(size_t)r * wa + c] : b[(size_t)r * wb + (c - wa)];
  dst[i] = f2bf(v);
}

__global__ void k_concat3(const float* __restrict__ a, int wa,
                          const float* __restrict__ b, int wb,
                          const float* __restrict__ c, int wc,
                          bf16_t* __restrict__ dst, int rows) {
  int w = wa + wb + wc;
  long i = (long)blockIdx.x * blockDim.x + threadIdx.x;
  if (i >= (long)rows * w) return;
  long r = i / w;
  int  col = (int)(i - r * w);
  float v;
  if (col < wa)            v = a[(size_t)r * wa + col];
  else if (col < wa + wb)  v = b[(size_t)r * wb + (col - wa)];
  else                     v = c[(size_t)r * wc + (col - wa - wb)];
  dst[i] = f2bf(v);
}

// softmax over S with key-padding mask; in-place fp32 + bf16 copy for next GEMM.
__global__ void k_softmax(float* __restrict__ sc, bf16_t* __restrict__ attnb,
                          const unsigned char* __restrict__ mask,
                          int NHv, int Tv, int Sv) {
  int row = blockIdx.x;                 // b*NH*T + h*T + t
  int b = row / (NHv * Tv);
  float* p = sc + (size_t)row * Sv;
  const unsigned char* mrow = mask + (size_t)b * Sv;
  __shared__ float red[256];

  float mx = -INFINITY;
  for (int s = threadIdx.x; s < Sv; s += blockDim.x) {
    float v = mrow[s] ? -INFINITY : p[s];
    mx = fmaxf(mx, v);
  }
  red[threadIdx.x] = mx; __syncthreads();
  for (int st = 128; st > 0; st >>= 1) {
    if (threadIdx.x < st) red[threadIdx.x] = fmaxf(red[threadIdx.x], red[threadIdx.x + st]);
    __syncthreads();
  }
  mx = red[0]; __syncthreads();

  float sum = 0.0f;
  for (int s = threadIdx.x; s < Sv; s += blockDim.x) {
    float v = mrow[s] ? 0.0f : expf(p[s] - mx);
    p[s] = v;
    sum += v;
  }
  red[threadIdx.x] = sum; __syncthreads();
  for (int st = 128; st > 0; st >>= 1) {
    if (threadIdx.x < st) red[threadIdx.x] += red[threadIdx.x + st];
    __syncthreads();
  }
  float inv = 1.0f / red[0];
  for (int s = threadIdx.x; s < Sv; s += blockDim.x) {
    float v = p[s] * inv;
    p[s] = v;
    attnb[(size_t)row * Sv + s] = f2bf(v);
  }
}

// PyTorch GRU cell gates: r,z,n split of gi/gh; h' = (1-z)*n + z*h. In-place h.
__global__ void k_gru_combine(const float* __restrict__ gi, long giStride,
                              const float* __restrict__ gh, long ghStride,
                              float* __restrict__ h, bf16_t* __restrict__ hbf,
                              float* __restrict__ seq, long seqStride,
                              int Bn, int Hn) {
  int i = blockIdx.x * blockDim.x + threadIdx.x;
  if (i >= Bn * Hn) return;
  int b = i / Hn, j = i - b * Hn;
  const float* gib = gi + (size_t)b * giStride;
  const float* ghb = gh + (size_t)b * ghStride;
  float ir = gib[j], iz = gib[Hn + j], in = gib[2 * Hn + j];
  float hr = ghb[j], hz = ghb[Hn + j], hn = ghb[2 * Hn + j];
  float r = 1.0f / (1.0f + expf(-(ir + hr)));
  float z = 1.0f / (1.0f + expf(-(iz + hz)));
  float n = tanhf(in + r * hn);
  float hp = h[(size_t)b * Hn + j];
  float ho = (1.0f - z) * n + z * hp;
  h[(size_t)b * Hn + j]   = ho;
  hbf[(size_t)b * Hn + j] = f2bf(ho);
  if (seq) seq[(size_t)b * seqStride + j] = ho;
}

__global__ void k_logsoftmax(float* __restrict__ x, int Vv) {
  int row = blockIdx.x;
  float* p = x + (size_t)row * Vv;
  __shared__ float red[256];

  float mx = -INFINITY;
  for (int i = threadIdx.x; i < Vv; i += blockDim.x) mx = fmaxf(mx, p[i]);
  red[threadIdx.x] = mx; __syncthreads();
  for (int st = 128; st > 0; st >>= 1) {
    if (threadIdx.x < st) red[threadIdx.x] = fmaxf(red[threadIdx.x], red[threadIdx.x + st]);
    __syncthreads();
  }
  mx = red[0]; __syncthreads();

  float sum = 0.0f;
  for (int i = threadIdx.x; i < Vv; i += blockDim.x) sum += expf(p[i] - mx);
  red[threadIdx.x] = sum; __syncthreads();
  for (int st = 128; st > 0; st >>= 1) {
    if (threadIdx.x < st) red[threadIdx.x] += red[threadIdx.x + st];
    __syncthreads();
  }
  float lse = logf(red[0]) + mx;
  for (int i = threadIdx.x; i < Vv; i += blockDim.x) p[i] = p[i] - lse;
}

// ---------------------------------------------------------------------------
// Host-side launch helpers
// ---------------------------------------------------------------------------
struct GemmB { int batches, nh; long sAb, sAh, sBb, sBh, sCb, sCh; };

static inline void gemm_tn(hipStream_t st, const bf16_t* A, const bf16_t* W, const float* bias,
                           float* C, bf16_t* Cbf, int M, int N, int K, int lda, int ldb, int ldc,
                           float alpha = 1.0f, int act = 0,
                           GemmB b = GemmB{1, 1, 0, 0, 0, 0, 0, 0}) {
  long tiles = (long)(M / 32) * (N / 64);            // 32x64 macro-tiles per wave
  dim3 grid((unsigned)((tiles + 7) / 8), (unsigned)b.batches, 1);
  k_wmma_gemm_tn<<<grid, 256, 0, st>>>(A, W, bias, C, Cbf, M, N, K, lda, ldb, ldc,
                                       alpha, act, b.nh, b.sAb, b.sAh, b.sBb, b.sBh, b.sCb, b.sCh);
}

static inline void gemm_nn(hipStream_t st, const bf16_t* A, const bf16_t* Bm,
                           float* C, bf16_t* Cbf, int M, int N, int K, int lda, int ldb, int ldc,
                           GemmB b = GemmB{1, 1, 0, 0, 0, 0, 0, 0}) {
  long tiles = (long)(M / 16) * (N / 16);
  dim3 grid((unsigned)((tiles + 7) / 8), (unsigned)b.batches, 1);
  k_wmma_gemm_nn<<<grid, 256, 0, st>>>(A, Bm, C, Cbf, M, N, K, lda, ldb, ldc,
                                       b.nh, b.sAb, b.sAh, b.sBb, b.sBh, b.sCb, b.sCh);
}

static inline unsigned gblk(long n) { return (unsigned)((n + 255) / 256); }

// ---------------------------------------------------------------------------
extern "C" void kernel_launch(void* const* d_in, const int* in_sizes, int n_in,
                              void* d_out, int out_size, void* d_ws, size_t ws_size,
                              hipStream_t stream) {
  (void)in_sizes; (void)n_in; (void)out_size; (void)ws_size;

  const int*   input_ids = (const int*)d_in[0];
  const float* enc_out   = (const float*)d_in[2];
  const float* enc_fin   = (const float*)d_in[3];
  const unsigned char* src_mask = (const unsigned char*)d_in[4];
  const float* emb_w     = (const float*)d_in[5];
  const float* inproj_w  = (const float*)d_in[6];
  const float* inproj_b  = (const float*)d_in[7];
  const float* outproj_w = (const float*)d_in[8];
  const float* outproj_b = (const float*)d_in[9];
  const float* bridge_w  = (const float*)d_in[10];
  const float* bridge_b  = (const float*)d_in[11];
  const float* wih0      = (const float*)d_in[12];
  const float* whh0      = (const float*)d_in[13];
  const float* bih0      = (const float*)d_in[14];
  const float* bhh0      = (const float*)d_in[15];
  const float* wih1      = (const float*)d_in[16];
  const float* whh1      = (const float*)d_in[17];
  const float* bih1      = (const float*)d_in[18];
  const float* bhh1      = (const float*)d_in[19];
  const float* preout_w  = (const float*)d_in[20];
  const float* outw      = (const float*)d_in[21];

  float* out_rnn  = (float*)d_out;                       // [B,T,H]
  float* out_hid  = out_rnn + (size_t)cB * cT * cH;      // [2,B,H]
  float* out_logp = out_hid + (size_t)2 * cB * cH;       // [B,T,V]

  // --- workspace bump allocator (256B aligned) ---
  char* wp = (char*)d_ws;
  auto alloc = [&](size_t bytes) -> char* {
    char* p = wp;
    wp += (bytes + 255) & ~(size_t)255;
    return p;
  };
  auto abf = [&](long n) { return (bf16_t*)alloc(sizeof(bf16_t) * (size_t)n); };
  auto af32 = [&](long n) { return (float*)alloc(sizeof(float) * (size_t)n); };

  bf16_t* enc_bf      = abf((long)cB * cS * cE);
  bf16_t* encfin_bf   = abf((long)cL * cB * cE);
  bf16_t* inproj_bf   = abf((long)3 * cE * cE);
  bf16_t* outprojw_bf = abf((long)cE * cE);
  bf16_t* bridgew_bf  = abf((long)cH * cE);
  bf16_t* wih0_bf     = abf((long)3 * cH * 2 * cE);
  bf16_t* whh0_bf     = abf((long)3 * cH * cH);
  bf16_t* wih1_bf     = abf((long)3 * cH * cH);
  bf16_t* whh1_bf     = abf((long)3 * cH * cH);
  bf16_t* preoutw_bf  = abf((long)cH * (2 * cE + cH));
  bf16_t* outw_bf     = abf((long)cV * cH);
  bf16_t* emb_bf      = abf((long)cB * cT * cE);
  float*  emb_f32     = af32((long)cB * cT * cE);
  bf16_t* q_bf        = abf((long)cB * cT * cE);
  bf16_t* k_bf        = abf((long)cB * cS * cE);
  bf16_t* v_bf        = abf((long)cB * cS * cE);
  float*  scores      = af32((long)cB * cNH * cT * cS);
  bf16_t* attn_bf     = abf((long)cB * cNH * cT * cS);
  bf16_t* ctx_bf      = abf((long)cB * cT * cE);
  float*  ctxp_f32    = af32((long)cB * cT * cE);
  bf16_t* ctxp_bf     = abf((long)cB * cT * cE);
  bf16_t* rnnin_bf    = abf((long)cB * cT * 2 * cE);
  float*  gi0_all     = af32((long)cB * cT * 3 * cH);
  float*  hid0        = af32((long)2 * cB * cH);
  bf16_t* h_bf        = abf((long)2 * cB * cH);
  float*  gh0         = af32((long)cB * 3 * cH);
  float*  gi1         = af32((long)cB * 3 * cH);
  float*  gh1         = af32((long)cB * 3 * cH);
  bf16_t* cat_bf      = abf((long)cB * cT * (2 * cE + cH));
  bf16_t* pre_bf      = abf((long)cB * cT * cH);

  auto cvt = [&](const float* s, bf16_t* d, long n) {
    k_f32_to_bf16<<<gblk(n), 256, 0, stream>>>(s, d, n);
  };

  // --- one-time fp32 -> bf16 weight/activation conversions (L2-resident reuse) ---
  cvt(enc_out,   enc_bf,      (long)cB * cS * cE);
  cvt(enc_fin,   encfin_bf,   (long)cL * cB * cE);
  cvt(inproj_w,  inproj_bf,   (long)3 * cE * cE);
  cvt(outproj_w, outprojw_bf, (long)cE * cE);
  cvt(bridge_w,  bridgew_bf,  (long)cH * cE);
  cvt(wih0,      wih0_bf,     (long)3 * cH * 2 * cE);
  cvt(whh0,      whh0_bf,     (long)3 * cH * cH);
  cvt(wih1,      wih1_bf,     (long)3 * cH * cH);
  cvt(whh1,      whh1_bf,     (long)3 * cH * cH);
  cvt(preout_w,  preoutw_bf,  (long)cH * (2 * cE + cH));
  cvt(outw,      outw_bf,     (long)cV * cH);

  // --- embedding gather ---
  k_embed<<<gblk((long)cB * cT * cE), 256, 0, stream>>>(input_ids, emb_w, emb_f32, emb_bf, cB * cT, cE);

  // --- Q/K/V projections (bias, bf16-out only) ---
  gemm_tn(stream, emb_bf, inproj_bf,                    inproj_b,          nullptr, q_bf, cB * cT, cE, cE, cE, cE, cE);
  gemm_tn(stream, enc_bf, inproj_bf + (size_t)cE * cE,  inproj_b + cE,     nullptr, k_bf, cB * cS, cE, cE, cE, cE, cE);
  gemm_tn(stream, enc_bf, inproj_bf + (size_t)2 * cE * cE, inproj_b + 2 * cE, nullptr, v_bf, cB * cS, cE, cE, cE, cE, cE);

  // --- scores[b,h] = q @ k^T / sqrt(DH), batched over B*NH ---
  {
    GemmB bb{cB * cNH, cNH, (long)cT * cE, (long)cDH, (long)cS * cE, (long)cDH,
             (long)cNH * cT * cS, (long)cT * cS};
    gemm_tn(stream, q_bf, k_bf, nullptr, scores, nullptr, cT, cS, cDH, cE, cE, cS, 0.125f, 0, bb);
  }

  // --- softmax (+mask) -> attn fp32 in place + bf16 ---
  k_softmax<<<cB * cNH * cT, 256, 0, stream>>>(scores, attn_bf, src_mask, cNH, cT, cS);

  // --- ctx[b,h] = attn @ v   (NN form), batched ---
  {
    GemmB bb{cB * cNH, cNH, (long)cNH * cT * cS, (long)cT * cS, (long)cS * cE, (long)cDH,
             (long)cT * cE, (long)cDH};
    gemm_nn(stream, attn_bf, v_bf, nullptr, ctx_bf, cT, cDH, cS, cS, cE, cE, bb);
  }

  // --- out_proj ---
  gemm_tn(stream, ctx_bf, outprojw_bf, outproj_b, ctxp_f32, ctxp_bf, cB * cT, cE, cE, cE, cE, cE);

  // --- rnn_in = concat(trg, ctx) -> bf16 ---
  k_concat2<<<gblk((long)cB * cT * 2 * cE), 256, 0, stream>>>(emb_f32, cE, ctxp_f32, cE, rnnin_bf, cB * cT);

  // --- bridge: hidden0 = tanh(encoder_final @ bridge_w^T + b) ---
  gemm_tn(stream, encfin_bf, bridgew_bf, bridge_b, hid0, h_bf, cL * cB, cH, cE, cE, cE, cH, 1.0f, /*tanh*/1);

  // --- precompute gi0 for all timesteps (teacher forcing => input known) ---
  gemm_tn(stream, rnnin_bf, wih0_bf, bih0, gi0_all, nullptr, cB * cT, 3 * cH, 2 * cE, 2 * cE, 2 * cE, 3 * cH);

  // --- sequential 2-layer GRU unroll ---
  float*  h0  = hid0;
  float*  h1  = hid0 + (size_t)cB * cH;
  bf16_t* h0b = h_bf;
  bf16_t* h1b = h_bf + (size_t)cB * cH;
  for (int t = 0; t < cT; ++t) {
    gemm_tn(stream, h0b, whh0_bf, bhh0, gh0, nullptr, cB, 3 * cH, cH, cH, cH, 3 * cH);
    k_gru_combine<<<gblk((long)cB * cH), 256, 0, stream>>>(
        gi0_all + (size_t)t * 3 * cH, (long)cT * 3 * cH, gh0, (long)3 * cH,
        h0, h0b, nullptr, 0, cB, cH);
    gemm_tn(stream, h0b, wih1_bf, bih1, gi1, nullptr, cB, 3 * cH, cH, cH, cH, 3 * cH);
    gemm_tn(stream, h1b, whh1_bf, bhh1, gh1, nullptr, cB, 3 * cH, cH, cH, cH, 3 * cH);
    k_gru_combine<<<gblk((long)cB * cH), 256, 0, stream>>>(
        gi1, (long)3 * cH, gh1, (long)3 * cH,
        h1, h1b, out_rnn + (size_t)t * cH, (long)cT * cH, cB, cH);
  }
  hipMemcpyAsync(out_hid, hid0, sizeof(float) * 2 * cB * cH, hipMemcpyDeviceToDevice, stream);

  // --- pre = concat(trg, rnn_out, ctx) @ pre_out_w^T ---
  k_concat3<<<gblk((long)cB * cT * (2 * cE + cH)), 256, 0, stream>>>(
      emb_f32, cE, out_rnn, cH, ctxp_f32, cE, cat_bf, cB * cT);
  gemm_tn(stream, cat_bf, preoutw_bf, nullptr, nullptr, pre_bf,
          cB * cT, cH, 2 * cE + cH, 2 * cE + cH, 2 * cE + cH, cH);

  // --- logits = pre @ out_w^T  (dominant GEMM: 2048 x 32000 x 1024) ---
  gemm_tn(stream, pre_bf, outw_bf, nullptr, out_logp, nullptr, cB * cT, cV, cH, cH, cH, cV);

  // --- log_softmax in place on d_out ---
  k_logsoftmax<<<cB * cT, 256, 0, stream>>>(out_logp, cV);
}
